// LSTM_49486613184738
// MI455X (gfx1250) — matrix operand order repaired and checked by reading, not compiled, not run on hardware
//
#include <hip/hip_runtime.h>
#include <hip/hip_bf16.h>

typedef __attribute__((ext_vector_type(16))) _Float16 v16h;
typedef __attribute__((ext_vector_type(8)))  float    v8f;
typedef __attribute__((ext_vector_type(4)))  unsigned int v4u;
typedef __attribute__((ext_vector_type(4)))  unsigned int u32x4;
typedef __attribute__((ext_vector_type(8)))  int          i32x8;
typedef __attribute__((ext_vector_type(4)))  int          i32x4;

// Problem constants (match reference)
#define B_   4096
#define T_   64
#define I_   158
#define IP_  160     // I padded to multiple of 32 (WMMA K)
#define H_   64
#define G4_  256     // 4*H
#define ROWS_ 16     // batch rows per workgroup (one WMMA M tile)
#define NTHR_ 128    // 4 waves

// ---- LDS partition (dynamic shared memory; dynamic block starts at offset 0) ----
constexpr int OFF_WIH  = 0;                      // _Float16 [G4][IP]
constexpr int SZ_WIH   = G4_ * IP_ * 2;          // 81920
constexpr int OFF_WHH  = OFF_WIH + SZ_WIH;       // _Float16 [G4][H]
constexpr int SZ_WHH   = G4_ * H_ * 2;           // 32768
constexpr int OFF_BIAS = OFF_WHH + SZ_WHH;       // float [G4]
constexpr int SZ_BIAS  = G4_ * 4;                // 1024
constexpr int OFF_XT   = OFF_BIAS + SZ_BIAS;     // _Float16 [ROWS][IP]
constexpr int SZ_XT    = ROWS_ * IP_ * 2;        // 5120
constexpr int OFF_GT   = OFF_XT + SZ_XT;         // float [ROWS][G4]
constexpr int SZ_GT    = ROWS_ * G4_ * 4;        // 16384
constexpr int OFF_HB   = OFF_GT + SZ_GT;         // _Float16 [ROWS][H]
constexpr int SZ_HB    = ROWS_ * H_ * 2;         // 2048
constexpr int OFF_CB   = OFF_HB + SZ_HB;         // float [ROWS][H]
constexpr int SZ_CB    = ROWS_ * H_ * 4;         // 4096
constexpr int OFF_STG  = OFF_CB + SZ_CB;         // float [ROWS][IP]  (TDM landing pad)
constexpr int SZ_STG   = ROWS_ * IP_ * 4;        // 10240
constexpr int SMEM_BYTES = OFF_STG + SZ_STG;     // 153600

union Frag16 { v16h v; v4u q[2]; };

// A-matrix fragment, 16-bit 16x32 layout (ISA 7.12.2):
// lane L holds row M=L%16; half j -> K = k0 + 16*(j/8) + 8*(L/16) + (j%8)
__device__ __forceinline__ v16h lds_fragA(const _Float16* row, int k0, int hi8) {
  Frag16 u;
  u.q[0] = *(const v4u*)(row + k0 + hi8);        // j = 0..7
  u.q[1] = *(const v4u*)(row + k0 + 16 + hi8);   // j = 8..15
  return u.v;
}

// B-matrix fragment (32x16): lane L holds col N=L%16, K = k0 + 16*(j/8) + 8*(L/16) + (j%8)
__device__ __forceinline__ v16h lds_fragB(const _Float16* row, int k0, int hi16) {
  Frag16 u;
  u.q[0] = *(const v4u*)(row + k0 + hi16 / 2);
  u.q[1] = *(const v4u*)(row + k0 + 16 + hi16 / 2);
  return u.v;
}

__device__ __forceinline__ float sigmoidf_(float v) {
  return 1.f / (1.f + __expf(-v));
}

__device__ __forceinline__ float leaky_(float s) {
  return (s >= 0.f) ? s : 0.01f * s;
}

// Issue a TDM 2D tile load: 16 rows x 160 f32 elements into LDS offset OFF_STG.
// tensor_dim0 = 158 -> columns 158..159 are OOB and land as zero (auto K-pad).
// Row stride = T*I elements (stride between consecutive batch rows at fixed t).
__device__ __forceinline__ void tdm_load_xtile(const float* gsrc) {
  unsigned long long ga = (unsigned long long)(uintptr_t)gsrc;
  u32x4 g0;
  g0[0] = 1u;                                    // count=1, user mode
  g0[1] = (unsigned)OFF_STG;                     // lds_addr (bytes, wave-relative)
  g0[2] = (unsigned)(ga & 0xffffffffu);          // global_addr[31:0]
  g0[3] = (unsigned)((ga >> 32) & 0x1ffffffu)    // global_addr[56:32]
        | (2u << 30);                            // type = 2 ("image")
  i32x8 g1;
  g1[0] = (int)(2u << 16);                       // data_size = 4B; no multicast/pad/iter
  g1[1] = (int)((unsigned)I_ << 16);             // tensor_dim0[15:0] = 158
  g1[2] = (int)((unsigned)ROWS_ << 16);          // tensor_dim1[15:0] = 16
  g1[3] = (int)((unsigned)IP_ << 16);            // tile_dim0 = 160
  g1[4] = (int)ROWS_;                            // tile_dim1 = 16, tile_dim2 = 0
  g1[5] = (int)(T_ * I_);                        // tensor_dim0_stride[31:0]
  g1[6] = 0;                                     // stride[47:32]=0, dim1_stride lo=0
  g1[7] = 0;
  i32x4 z4 = {0, 0, 0, 0};
#if __clang_major__ >= 23
  i32x8 z8 = {0, 0, 0, 0, 0, 0, 0, 0};
  __builtin_amdgcn_tensor_load_to_lds(g0, g1, z4, z4, z8, 0);
#else
  __builtin_amdgcn_tensor_load_to_lds(g0, g1, z4, z4, 0);
#endif
}

extern "C" __global__ __launch_bounds__(NTHR_)
__attribute__((amdgpu_waves_per_eu(2)))
void lstm_fused_wmma(const float* __restrict__ x,
                     const float* __restrict__ W_ih, const float* __restrict__ W_hh,
                     const float* __restrict__ b_ih, const float* __restrict__ b_hh,
                     const float* __restrict__ W1, const float* __restrict__ b1,
                     const float* __restrict__ W2, const float* __restrict__ b2,
                     const float* __restrict__ W3, const float* __restrict__ b3,
                     float* __restrict__ out) {
  extern __shared__ char smem[];
  _Float16* Wb_ih = (_Float16*)(smem + OFF_WIH);
  _Float16* Wb_hh = (_Float16*)(smem + OFF_WHH);
  float*    biasv = (float*)   (smem + OFF_BIAS);
  _Float16* xt    = (_Float16*)(smem + OFF_XT);
  float*    gates = (float*)   (smem + OFF_GT);
  _Float16* hbuf  = (_Float16*)(smem + OFF_HB);
  float*    cbuf  = (float*)   (smem + OFF_CB);
  float*    stage = (float*)   (smem + OFF_STG);

  const int tid  = threadIdx.x;
  const int lane = tid & 31;
  const int wave = tid >> 5;           // 0..3 -> gate i,f,g,o column range
  const int b0   = blockIdx.x * ROWS_;

  // Kick off TDM load of x_{t=0} tile while we stage weights.
  if (wave == 0)
    tdm_load_xtile(&x[(size_t)b0 * T_ * I_]);

  // ---- one-time staging: weights -> f16 LDS (B-fragment friendly), bias, state ----
  for (int i = tid; i < G4_ * IP_; i += NTHR_) {
    int g = i / IP_, k = i % IP_;
    Wb_ih[i] = (k < I_) ? (_Float16)W_ih[g * I_ + k] : (_Float16)0.f;
  }
  for (int i = tid; i < G4_ * H_; i += NTHR_) Wb_hh[i] = (_Float16)W_hh[i];
  for (int g = tid; g < G4_; g += NTHR_) biasv[g] = b_ih[g] + b_hh[g];
  for (int i = tid; i < ROWS_ * H_; i += NTHR_) { cbuf[i] = 0.f; hbuf[i] = (_Float16)0.f; }

  const int mrow    = lane & 15;         // A row / B-C column within tile
  const int hi8     = (lane >> 4) * 8;   // A fragment K sub-offset
  const int hi16    = (lane >> 4) * 16;  // B fragment K sub-offset
  const int colbase = wave * 64;         // this wave's gate columns

  for (int t = 0; t < T_; ++t) {
    // Wait for the TDM tile issued last iteration, then publish to all waves.
    if (wave == 0) __builtin_amdgcn_s_wait_tensorcnt(0);
    __syncthreads();

    // Convert staged f32 tile -> f16 xt (OOB columns already zeroed by TDM).
    for (int i = tid; i < ROWS_ * IP_; i += NTHR_)
      xt[i] = (_Float16)stage[i];
    __syncthreads();                      // xt ready; stage reusable

    // Prefetch next timestep's tile via TDM, overlapped with WMMA below.
    if (wave == 0 && t + 1 < T_)
      tdm_load_xtile(&x[((size_t)b0 * T_ + (t + 1)) * I_]);

    // gates[:, colbase:colbase+64] = x_t @ W_ih^T + h @ W_hh^T   (bias added later)
    v8f acc[4] = {};
    const _Float16* arow = xt + mrow * IP_;
    #pragma unroll
    for (int ks = 0; ks < IP_ / 32; ++ks) {          // K over input features
      v16h a = lds_fragA(arow, ks * 32, hi8);
      #pragma unroll
      for (int n = 0; n < 4; ++n) {
        const _Float16* brow = Wb_ih + (size_t)(colbase + n * 16 + mrow) * IP_;
        v16h bf = lds_fragB(brow, ks * 32, hi16);
        acc[n] = __builtin_amdgcn_wmma_f32_16x16x32_f16(
            false, a, false, bf, (short)0, acc[n], false, false);
      }
    }
    const _Float16* hrow = hbuf + mrow * H_;
    #pragma unroll
    for (int ks = 0; ks < H_ / 32; ++ks) {           // K over hidden state
      v16h a = lds_fragA(hrow, ks * 32, hi8);
      #pragma unroll
      for (int n = 0; n < 4; ++n) {
        const _Float16* brow = Wb_hh + (size_t)(colbase + n * 16 + mrow) * H_;
        v16h bf = lds_fragB(brow, ks * 32, hi16);
        acc[n] = __builtin_amdgcn_wmma_f32_16x16x32_f16(
            false, a, false, bf, (short)0, acc[n], false, false);
      }
    }
    // C/D layout: VGPR v, lanes<16 -> M=v, lanes>=16 -> M=v+8 ; N = lane%16
    #pragma unroll
    for (int n = 0; n < 4; ++n)
      #pragma unroll
      for (int v = 0; v < 8; ++v)
        gates[(v + (lane >> 4) * 8) * G4_ + colbase + n * 16 + mrow] = acc[n][v];
    __syncthreads();

    // LSTM cell update (bias folded in here; 128 threads over 16x64 cells)
    for (int e = tid; e < ROWS_ * H_; e += NTHR_) {
      int r = e >> 6, c = e & 63;
      float iv = sigmoidf_(gates[r * G4_ + c]         + biasv[c]);
      float fv = sigmoidf_(gates[r * G4_ + 64 + c]    + biasv[64 + c]);
      float gv = tanhf   (gates[r * G4_ + 128 + c]    + biasv[128 + c]);
      float ov = sigmoidf_(gates[r * G4_ + 192 + c]   + biasv[192 + c]);
      float cc = fv * cbuf[e] + iv * gv;
      cbuf[e] = cc;
      hbuf[e] = (_Float16)(ov * tanhf(cc));
    }
    __syncthreads();
  }

  // ================= MLP head 64->32->16->1 via WMMA =================
  // Recycle dead LDS: W1/W2 (f16) into the old Wb_ih region; activations
  // into xt (f16) and gates (f32); b1/b2 into biasv.
  _Float16* mW1   = (_Float16*)(smem + OFF_WIH);          // [32][64] f16
  _Float16* mW2   = (_Float16*)(smem + OFF_WIH + 4096);   // [16][32] f16
  _Float16* l1buf = xt;                                   // [16][32] f16
  float*    l2buf = gates;                                // [16][16] f32

  for (int i = tid; i < 32 * H_; i += NTHR_) mW1[i] = (_Float16)W1[i];
  for (int i = tid; i < 16 * 32; i += NTHR_) mW2[i] = (_Float16)W2[i];
  if (tid < 32)      biasv[tid] = b1[tid];
  else if (tid < 48) biasv[tid] = b2[tid - 32];
  __syncthreads();

  // Layer 1: [16x64] @ [64x32] + b1, LeakyReLU. Waves 0,1: one 16-col tile each.
  if (wave < 2) {
    v8f acc = {};
    const _Float16* arow = hbuf + mrow * H_;
    #pragma unroll
    for (int ks = 0; ks < 2; ++ks) {
      v16h a  = lds_fragA(arow, ks * 32, hi8);
      v16h bf = lds_fragB(mW1 + (size_t)(wave * 16 + mrow) * H_, ks * 32, hi16);
      acc = __builtin_amdgcn_wmma_f32_16x16x32_f16(
          false, a, false, bf, (short)0, acc, false, false);
    }
    const int col = wave * 16 + mrow;
    const float bb = biasv[col];
    #pragma unroll
    for (int v = 0; v < 8; ++v)
      l1buf[(v + (lane >> 4) * 8) * 32 + col] = (_Float16)leaky_(acc[v] + bb);
  }
  __syncthreads();

  // Layer 2: [16x32] @ [32x16] + b2, LeakyReLU. One WMMA (K=32), wave 0.
  if (wave == 0) {
    v8f acc = {};
    v16h a  = lds_fragA(l1buf + mrow * 32, 0, hi8);
    v16h bf = lds_fragB(mW2 + (size_t)mrow * 32, 0, hi16);
    acc = __builtin_amdgcn_wmma_f32_16x16x32_f16(
        false, a, false, bf, (short)0, acc, false, false);
    const float bb = biasv[32 + mrow];
    #pragma unroll
    for (int v = 0; v < 8; ++v)
      l2buf[(v + (lane >> 4) * 8) * 16 + mrow] = leaky_(acc[v] + bb);
  }
  __syncthreads();

  // Layer 3: [16x16] @ [16x1] + b3. Lanes 0..15 each finish one batch row.
  if (tid < ROWS_) {
    float s = b3[0];
    #pragma unroll
    for (int k = 0; k < 16; ++k) s += W3[k] * l2buf[tid * 16 + k];
    out[b0 + tid] = s;
  }
}

extern "C" void kernel_launch(void* const* d_in, const int* in_sizes, int n_in,
                              void* d_out, int out_size, void* d_ws, size_t ws_size,
                              hipStream_t stream) {
  (void)in_sizes; (void)n_in; (void)d_ws; (void)ws_size; (void)out_size;
  const float* x    = (const float*)d_in[0];
  const float* W_ih = (const float*)d_in[1];
  const float* W_hh = (const float*)d_in[2];
  const float* b_ih = (const float*)d_in[3];
  const float* b_hh = (const float*)d_in[4];
  const float* W1   = (const float*)d_in[5];
  const float* b1   = (const float*)d_in[6];
  const float* W2   = (const float*)d_in[7];
  const float* b2   = (const float*)d_in[8];
  const float* W3   = (const float*)d_in[9];
  const float* b3   = (const float*)d_in[10];
  float* out = (float*)d_out;

  // Allow >64KB dynamic LDS (gfx1250 WGP has 320KB).
  (void)hipFuncSetAttribute((const void*)lstm_fused_wmma,
                            hipFuncAttributeMaxDynamicSharedMemorySize, SMEM_BYTES);

  dim3 grid(B_ / ROWS_), block(NTHR_);
  lstm_fused_wmma<<<grid, block, SMEM_BYTES, stream>>>(
      x, W_ih, W_hh, b_ih, b_hh, W1, b1, W2, b2, W3, b3, out);
}
